// Model_11570641896139
// MI455X (gfx1250) — compile-verified, matrix operand order checked by reading, not run on hardware
//
#include <hip/hip_runtime.h>
#include <hip/hip_bf16.h>

// ---------------------------------------------------------------------------
// MI455X (gfx1250). sparse_code() in the reference is dead code -> skipped.
// All tensors channel-contiguous (B, T, C) so WMMA f16 fragments load as
// two 128-bit vectors per lane (fragment K-runs are contiguous).
// ---------------------------------------------------------------------------

typedef __attribute__((ext_vector_type(16))) _Float16 v16h;
typedef __attribute__((ext_vector_type(8)))  _Float16 v8h;
typedef __attribute__((ext_vector_type(8)))  float    v8f;

// f16 A/B fragment for 16x16x32: lane's elements are K = base0+0..7 and
// base0+16..23 with base0 = (lane&16) ? 8 : 0  (ISA 7.12.2). With K
// contiguous in memory this is two aligned b128 loads.
__device__ __forceinline__ v16h load_frag(const _Float16* kbase, int lane) {
    const _Float16* p = kbase + ((lane & 16) ? 8 : 0);
    union { v16h v; v8h h[2]; } u;
    u.h[0] = *(const v8h*)(p);
    u.h[1] = *(const v8h*)(p + 16);
    return u.v;
}

__device__ __forceinline__ v16h zero_frag() {
    v16h z;
#pragma unroll
    for (int e = 0; e < 16; ++e) z[e] = (_Float16)0.0f;
    return z;
}

// ---------------------------------------------------------------------------
// 1) embed: e[(b,band,t), f] = x row . embed_w col + b ; write encoded in
//    (B, T, C=1024) layout, fp32 (ctx path) + f16 (conv GEMM A).
//    channel c = f*128 + band.
// ---------------------------------------------------------------------------
__global__ __launch_bounds__(256)
void embed_kernel(const float* __restrict__ x,     // (65536, 257)
                  const float* __restrict__ ew,    // (257, 8)
                  const float* __restrict__ eb,    // (8)
                  float* __restrict__ enc_f32,     // (B, T, 1024)
                  _Float16* __restrict__ enc_f16)  // (B, T, 1024)
{
    const int wave = threadIdx.x >> 5;
    const int lane = threadIdx.x & 31;
    const int row  = blockIdx.x * 8 + wave;        // (b, band, t) flat
    const float* xr = x + (size_t)row * 257;

    float acc[8] = {0.f, 0.f, 0.f, 0.f, 0.f, 0.f, 0.f, 0.f};
    for (int k = lane; k < 257; k += 32) {
        float xv = xr[k];
        const float* wk = ew + k * 8;
#pragma unroll
        for (int f = 0; f < 8; ++f) acc[f] += xv * wk[f];
    }
#pragma unroll
    for (int f = 0; f < 8; ++f) {
#pragma unroll
        for (int off = 16; off > 0; off >>= 1)
            acc[f] += __shfl_xor(acc[f], off, 32);
    }
    if (lane < 8) {
        int b    = row >> 14;
        int band = (row >> 7) & 127;
        int t    = row & 127;
        float e  = acc[lane] + eb[lane];
        size_t idx = (((size_t)(b * 128 + t)) << 10) + (size_t)(lane * 128 + band);
        enc_f32[idx] = e;
        enc_f16[idx] = (_Float16)e;
    }
}

// ctx[b, c] = sum_t enc[b, t, c]   (coalesced across c)
__global__ __launch_bounds__(256)
void ctx_reduce_kernel(const float* __restrict__ enc, float* __restrict__ ctx)
{
    int i = blockIdx.x * 256 + threadIdx.x;   // 0..4095 = b*1024 + c
    if (i >= 4096) return;
    int b = i >> 10, c = i & 1023;
    const float* p = enc + (((size_t)b * 128) << 10) + c;
    float s = 0.f;
    for (int t = 0; t < 128; ++t) s += p[(size_t)t << 10];
    ctx[i] = s;
}

// ---------------------------------------------------------------------------
// Weight conversions.
// ---------------------------------------------------------------------------
// ctx weights (K,N) fp32 -> (N,K) f16 so B-fragment K-runs are contiguous.
__global__ void cvt_ctxw_kernel(const float* __restrict__ src,  // (1024,1024) [k][n]
                                _Float16* __restrict__ dst)     // (1024,1024) [n][k]
{
    int i = blockIdx.x * 256 + threadIdx.x;      // coalesced read index k*1024+n
    if (i >= 1024 * 1024) return;
    int k = i >> 10, n = i & 1023;
    dst[((size_t)n << 10) + k] = (_Float16)src[i];
}

// conv weights (Cout,Cin,7) fp32 -> (7,Cout,Cin) f16 (K=Cin contiguous).
__global__ void cvt_convw_kernel(const float* __restrict__ src,
                                 _Float16* __restrict__ dst,
                                 int Cout, int Cin)
{
    int i = blockIdx.x * 256 + threadIdx.x;
    int total = Cout * Cin * 7;
    if (i >= total) return;
    int co = i / (Cin * 7);
    int r  = i - co * (Cin * 7);
    int ci = r / 7;
    int kk = r - ci * 7;
    dst[((size_t)kk * Cout + co) * Cin + ci] = (_Float16)src[i];
}

// ---------------------------------------------------------------------------
// 2) ctx dense layer (single WG, 16 waves, M padded 4->16, WMMA).
// ---------------------------------------------------------------------------
#define CTX_N 1024
__global__ __launch_bounds__(512)
void dense_ctx_kernel(const float* __restrict__ hin,    // (4,1024)
                      const _Float16* __restrict__ Wt,  // (1024,1024) [n][k]
                      const float* __restrict__ bias,
                      const float* __restrict__ gamma,
                      const float* __restrict__ beta,
                      float* __restrict__ hout,         // (4,1024)
                      int do_ln)
{
    __shared__ __align__(16) _Float16 Ash[16 * CTX_N];
    __shared__ float hbuf[4 * CTX_N];
    __shared__ float red1[512];
    __shared__ float red2[512];
    __shared__ float s_mean, s_rstd;

    const int tid = threadIdx.x;
    for (int i = tid; i < 16 * CTX_N; i += 512) {
        int m = i >> 10, k = i & (CTX_N - 1);
        Ash[i] = (m < 4) ? (_Float16)hin[m * CTX_N + k] : (_Float16)0.0f;
    }
    __syncthreads();

    const int wave = tid >> 5;
    const int lane = tid & 31;
    const int mA   = lane & 15;

    v8f acc0 = {}, acc1 = {}, acc2 = {}, acc3 = {};
    for (int kc = 0; kc < CTX_N; kc += 32) {
        v16h a = load_frag(&Ash[mA * CTX_N + kc], lane);
        const int nb = wave * 64 + (lane & 15);
        v16h b0 = load_frag(Wt + ((size_t)(nb)      << 10) + kc, lane);
        v16h b1 = load_frag(Wt + ((size_t)(nb + 16) << 10) + kc, lane);
        v16h b2 = load_frag(Wt + ((size_t)(nb + 32) << 10) + kc, lane);
        v16h b3 = load_frag(Wt + ((size_t)(nb + 48) << 10) + kc, lane);
        acc0 = __builtin_amdgcn_wmma_f32_16x16x32_f16(false, a, false, b0, (short)0, acc0, false, false);
        acc1 = __builtin_amdgcn_wmma_f32_16x16x32_f16(false, a, false, b1, (short)0, acc1, false, false);
        acc2 = __builtin_amdgcn_wmma_f32_16x16x32_f16(false, a, false, b2, (short)0, acc2, false, false);
        acc3 = __builtin_amdgcn_wmma_f32_16x16x32_f16(false, a, false, b3, (short)0, acc3, false, false);
    }
    if (lane < 16) {                 // D: lanes 0..15 hold M = vgpr index
#pragma unroll
        for (int v = 0; v < 4; ++v) {
            hbuf[v * CTX_N + wave * 64 + lane +  0] = acc0[v];
            hbuf[v * CTX_N + wave * 64 + lane + 16] = acc1[v];
            hbuf[v * CTX_N + wave * 64 + lane + 32] = acc2[v];
            hbuf[v * CTX_N + wave * 64 + lane + 48] = acc3[v];
        }
    }
    __syncthreads();

    for (int i = tid; i < 4 * CTX_N; i += 512) {
        int nn = i & (CTX_N - 1);
        float vv = hbuf[i] + bias[nn];
        if (do_ln) vv = (vv > 0.f) ? vv : 0.2f * vv;
        hbuf[i] = vv;
    }
    __syncthreads();

    if (do_ln) {
        for (int m = 0; m < 4; ++m) {
            float s = 0.f, s2 = 0.f;
            for (int nn = tid; nn < CTX_N; nn += 512) {
                float vv = hbuf[m * CTX_N + nn];
                s += vv; s2 += vv * vv;
            }
            red1[tid] = s; red2[tid] = s2;
            __syncthreads();
            for (int off = 256; off > 0; off >>= 1) {
                if (tid < off) { red1[tid] += red1[tid + off]; red2[tid] += red2[tid + off]; }
                __syncthreads();
            }
            if (tid == 0) {
                float mean = red1[0] * (1.0f / CTX_N);
                float var  = red2[0] * (1.0f / CTX_N) - mean * mean;
                s_mean = mean;
                s_rstd = rsqrtf(var + 1e-5f);
            }
            __syncthreads();
            float mean = s_mean, rstd = s_rstd;
            for (int nn = tid; nn < CTX_N; nn += 512)
                hbuf[m * CTX_N + nn] =
                    (hbuf[m * CTX_N + nn] - mean) * rstd * gamma[nn] + beta[nn];
            __syncthreads();
        }
    }
    for (int i = tid; i < 4 * CTX_N; i += 512)
        hout[i] = hbuf[i];
}

// ---------------------------------------------------------------------------
// 3) conv1d(k=7,pad=3) implicit GEMM. One wave computes 16(M) x 16*NT(N);
//    A fragment shared across the NT accumulators. (B,T,C) layouts.
// ---------------------------------------------------------------------------
template <int NT>
__global__ __launch_bounds__(32)
void conv1d_wmma_kernel(const _Float16* __restrict__ in,  // (B, T, Cin)
                        const _Float16* __restrict__ w,   // (7, Cout, Cin)
                        const float* __restrict__ bias,   // (Cout)
                        float* __restrict__ out,          // (B, T, Cout)
                        int Cin, int Cout, int T)
{
    const int lane = threadIdx.x;
    const int m0   = blockIdx.x * 16;
    const int n0   = blockIdx.y * (16 * NT);
    const int mrow = m0 + (lane & 15);
    const int b_a  = mrow / T;
    const int t_a  = mrow - b_a * T;

    v8f acc[NT] = {};
#pragma unroll
    for (int kk = 0; kk < 7; ++kk) {
        const int tt  = t_a + kk - 3;
        const bool tin = (tt >= 0) && (tt < T);
        const _Float16* arow = in + ((size_t)(b_a * T + tt)) * Cin;
        const _Float16* wrow = w + (size_t)kk * Cout * Cin;
        for (int kc = 0; kc < Cin; kc += 32) {
            v16h a = tin ? load_frag(arow + kc, lane) : zero_frag();
            if (tin) __builtin_prefetch(arow + kc + 64, 0, 0);
#pragma unroll
            for (int t2 = 0; t2 < NT; ++t2) {
                int n = n0 + t2 * 16 + (lane & 15);
                v16h bf = (n < Cout) ? load_frag(wrow + (size_t)n * Cin + kc, lane)
                                     : zero_frag();
                acc[t2] = __builtin_amdgcn_wmma_f32_16x16x32_f16(
                    false, a, false, bf, (short)0, acc[t2], false, false);
            }
        }
    }
#pragma unroll
    for (int t2 = 0; t2 < NT; ++t2) {
        int n = n0 + t2 * 16 + (lane & 15);
        if (n < Cout) {
            float bv = bias[n];
#pragma unroll
            for (int v = 0; v < 8; ++v) {
                int row = m0 + v + ((lane & 16) ? 8 : 0);
                out[(size_t)row * Cout + n] = acc[t2][v] + bv;
            }
        }
    }
}

// ---------------------------------------------------------------------------
// BatchNorm stats: deterministic two-stage reduction over BT = 4*T.
// Stage A: 64 fixed chunks, coalesced (channels across lanes).
// ---------------------------------------------------------------------------
#define BN_CHUNKS 64
__global__ __launch_bounds__(256)
void bn_partial_kernel(const float* __restrict__ src, int C, int BT,
                       float* __restrict__ ps, float* __restrict__ ps2)
{
    int c = blockIdx.x * 256 + threadIdx.x;
    int chunk = blockIdx.y;
    if (c >= C) return;
    int len = BT / BN_CHUNKS;
    int beg = chunk * len;
    float s = 0.f, s2 = 0.f;
    for (int i = 0; i < len; ++i) {
        float v = src[(size_t)(beg + i) * C + c];
        s += v; s2 += v * v;
    }
    ps [chunk * C + c] = s;
    ps2[chunk * C + c] = s2;
}

__global__ __launch_bounds__(256)
void bn_final_kernel(const float* __restrict__ ps, const float* __restrict__ ps2,
                     int C, int BT,
                     float* __restrict__ mean, float* __restrict__ rstd)
{
    int c = blockIdx.x * 256 + threadIdx.x;
    if (c >= C) return;
    float s = 0.f, s2 = 0.f;
    for (int k = 0; k < BN_CHUNKS; ++k) { s += ps[k * C + c]; s2 += ps2[k * C + c]; }
    float m   = s / (float)BT;
    float var = s2 / (float)BT - m * m;
    mean[c] = m;
    rstd[c] = rsqrtf(var + 1e-5f);
}

// fused bn + leaky_relu(0.2) + upsample4 : (B,T,C) fp32 -> (B,4T,C) f16
__global__ __launch_bounds__(256)
void bn_apply_up4_kernel(const float* __restrict__ src, int C, int T,
                         const float* __restrict__ mean, const float* __restrict__ rstd,
                         const float* __restrict__ g, const float* __restrict__ bb,
                         _Float16* __restrict__ dst)
{
    size_t idx = (size_t)blockIdx.x * 256 + threadIdx.x;   // over (b, t, c)
    size_t total = (size_t)4 * T * C;
    if (idx >= total) return;
    int    c  = (int)(idx % (size_t)C);
    size_t bt = idx / (size_t)C;
    size_t b  = bt / (size_t)T;
    size_t t  = bt % (size_t)T;
    float v = src[idx];
    v = (v - mean[c]) * rstd[c] * g[c] + bb[c];
    v = (v > 0.f) ? v : 0.2f * v;
    _Float16 h = (_Float16)v;
    size_t base = ((b * 4 * T) + 4 * t) * (size_t)C + c;
    dst[base]         = h;
    dst[base + C]     = h;
    dst[base + 2 * C] = h;
    dst[base + 3 * C] = h;
}

// ---------------------------------------------------------------------------
// Launch
// ---------------------------------------------------------------------------
extern "C" void kernel_launch(void* const* d_in, const int* in_sizes, int n_in,
                              void* d_out, int out_size, void* d_ws, size_t ws_size,
                              hipStream_t stream)
{
    (void)in_sizes; (void)n_in; (void)out_size; (void)ws_size;

    const float* x  = (const float*)d_in[0];
    const float* ew = (const float*)d_in[2];
    const float* eb = (const float*)d_in[3];
    const float* cw[3]  = { (const float*)d_in[4],  (const float*)d_in[8],  (const float*)d_in[12] };
    const float* cb[3]  = { (const float*)d_in[5],  (const float*)d_in[9],  (const float*)d_in[13] };
    const float* cg[2]  = { (const float*)d_in[6],  (const float*)d_in[10] };
    const float* cbe[2] = { (const float*)d_in[7],  (const float*)d_in[11] };
    const float* uw[5]  = { (const float*)d_in[14], (const float*)d_in[18], (const float*)d_in[22],
                            (const float*)d_in[26], (const float*)d_in[30] };
    const float* ub[5]  = { (const float*)d_in[15], (const float*)d_in[19], (const float*)d_in[23],
                            (const float*)d_in[27], (const float*)d_in[31] };
    const float* bg[4]  = { (const float*)d_in[16], (const float*)d_in[20], (const float*)d_in[24],
                            (const float*)d_in[28] };
    const float* bb[4]  = { (const float*)d_in[17], (const float*)d_in[21], (const float*)d_in[25],
                            (const float*)d_in[29] };

    char* ws = (char*)d_ws;
    size_t off = 0;
    auto take = [&](size_t bytes) -> void* {
        void* p = ws + off;
        off = (off + bytes + 255) & ~(size_t)255;
        return p;
    };
    float*    enc_f32 = (float*)   take((size_t)4 * 128 * 1024 * 4);
    _Float16* enc_f16 = (_Float16*)take((size_t)4 * 128 * 1024 * 2);
    float*    ctx0    = (float*)   take(4096 * 4);
    float*    h1      = (float*)   take(4096 * 4);
    float*    h2      = (float*)   take(4096 * 4);
    _Float16* cwf[3];
    for (int i = 0; i < 3; ++i) cwf[i] = (_Float16*)take((size_t)1024 * 1024 * 2);
    const int convCout[5] = {512, 256, 128, 64, 1};
    const int convCin[5]  = {1024, 512, 256, 128, 64};
    _Float16* uwT[5];
    for (int i = 0; i < 5; ++i)
        uwT[i] = (_Float16*)take((size_t)convCout[i] * convCin[i] * 7 * 2);
    float*    out1 = (float*)   take((size_t)4 * 128  * 512 * 4);
    _Float16* in2  = (_Float16*)take((size_t)4 * 512  * 512 * 2);
    float*    out2 = (float*)   take((size_t)4 * 512  * 256 * 4);
    _Float16* in3  = (_Float16*)take((size_t)4 * 2048 * 256 * 2);
    float*    out3 = (float*)   take((size_t)4 * 2048 * 128 * 4);
    _Float16* in4  = (_Float16*)take((size_t)4 * 8192 * 128 * 2);
    float*    out4 = (float*)   take((size_t)4 * 8192 * 64  * 4);
    _Float16* in5  = (_Float16*)take((size_t)4 * 32768 * 64 * 2);
    float*    meanb = (float*)take(512 * 4);
    float*    rstdb = (float*)take(512 * 4);
    float*    ps    = (float*)take((size_t)BN_CHUNKS * 512 * 4);
    float*    ps2   = (float*)take((size_t)BN_CHUNKS * 512 * 4);

    float* y_out   = (float*)d_out;            // (4, 32768) == (B,T,1)
    float* ctx_out = (float*)d_out + 131072;   // (4, 1024)

    // weight conversions
    for (int i = 0; i < 3; ++i)
        cvt_ctxw_kernel<<<(1024 * 1024 + 255) / 256, 256, 0, stream>>>(cw[i], cwf[i]);
    for (int i = 0; i < 5; ++i) {
        int total = convCout[i] * convCin[i] * 7;
        cvt_convw_kernel<<<(total + 255) / 256, 256, 0, stream>>>(uw[i], uwT[i],
                                                                  convCout[i], convCin[i]);
    }

    // embed + ctx
    embed_kernel<<<8192, 256, 0, stream>>>(x, ew, eb, enc_f32, enc_f16);
    ctx_reduce_kernel<<<16, 256, 0, stream>>>(enc_f32, ctx0);
    dense_ctx_kernel<<<1, 512, 0, stream>>>(ctx0, cwf[0], cb[0], cg[0], cbe[0], h1, 1);
    dense_ctx_kernel<<<1, 512, 0, stream>>>(h1,   cwf[1], cb[1], cg[1], cbe[1], h2, 1);
    dense_ctx_kernel<<<1, 512, 0, stream>>>(h2,   cwf[2], cb[2], cb[2], cb[2], ctx_out, 0);

    // conv / bn / upsample chain
    const _Float16* cin[5]   = { enc_f16, in2, in3, in4, in5 };
    float*          cout_[5] = { out1, out2, out3, out4, y_out };
    _Float16*       nxt[4]   = { in2, in3, in4, in5 };
    const int T[5] = {128, 512, 2048, 8192, 32768};

    for (int s = 0; s < 5; ++s) {
        int M = 4 * T[s];
        if (s < 4) {
            dim3 grid(M / 16, convCout[s] / 64);
            conv1d_wmma_kernel<4><<<grid, 32, 0, stream>>>(cin[s], uwT[s], ub[s], cout_[s],
                                                           convCin[s], convCout[s], T[s]);
            int BT = 4 * T[s];
            dim3 sgrid((convCout[s] + 255) / 256, BN_CHUNKS);
            bn_partial_kernel<<<sgrid, 256, 0, stream>>>(cout_[s], convCout[s], BT, ps, ps2);
            bn_final_kernel<<<(convCout[s] + 255) / 256, 256, 0, stream>>>(ps, ps2,
                                                                           convCout[s], BT,
                                                                           meanb, rstdb);
            size_t total = (size_t)BT * convCout[s];
            bn_apply_up4_kernel<<<(unsigned)((total + 255) / 256), 256, 0, stream>>>(
                cout_[s], convCout[s], T[s], meanb, rstdb, bg[s], bb[s], nxt[s]);
        } else {
            dim3 grid(M / 16, 1);
            conv1d_wmma_kernel<1><<<grid, 32, 0, stream>>>(cin[s], uwT[s], ub[s], cout_[s],
                                                           convCin[s], convCout[s], T[s]);
        }
    }
}
